// FullMHA_5471788335677
// MI455X (gfx1250) — compile-verified, matrix operand order checked by reading, not run on hardware
//
#include <hip/hip_runtime.h>

// ---------- types ----------
typedef __bf16 bf16;
typedef __attribute__((ext_vector_type(16))) __bf16        v16bf;
typedef __attribute__((ext_vector_type(8)))  float         v8f;
typedef __attribute__((ext_vector_type(4)))  unsigned int  u32x4;
typedef __attribute__((ext_vector_type(2)))  unsigned int  u32x2;
typedef __attribute__((ext_vector_type(4)))  float         f32x4;

#define WMMA_BF16(a, b, c) \
  __builtin_amdgcn_wmma_f32_16x16x32_bf16(false, (a), false, (b), (short)0, (c), false, false)

// Dimensions (fixed by the reference)
#define BD   32
#define TD   512
#define CD   2048
#define HD   16
#define DD   128

// ---------- helpers ----------
// Load a 16x32 bf16 A/B fragment from row-major (K-contiguous) storage.
// Lane l: row = caller-provided (includes lane&15); K layout: vgprs0-3 hold
// K = half*8 + {0..7}, vgprs4-7 hold K = 16 + half*8 + {0..7}.
__device__ __forceinline__ v16bf lds_frag(const bf16* base, int row, int ldk, int lane) {
  const bf16* p = base + row * ldk + ((lane >> 4) << 3);
  union { u32x4 u[2]; v16bf v; } r;
  r.u[0] = *(const u32x4*)(p);
  r.u[1] = *(const u32x4*)(p + 16);
  return r.v;
}

__device__ __forceinline__ u32x2 pack4_bf16(f32x4 f) {
  union { bf16 h[4]; u32x2 u; } r;
  r.h[0] = (bf16)f.x; r.h[1] = (bf16)f.y; r.h[2] = (bf16)f.z; r.h[3] = (bf16)f.w;
  return r.u;
}

// =====================================================================
// Kernel 1: QKV projection.  qkv[bt, f] = sum_c x[bt,c] * w_qkv[f,c]
// Both operands are K(=c)-major -> identical fragment loads.
// Tile: BM=128 x BN=64 x BK=32, 256 threads (8 waves, each 32x32 out).
// Epilogue scatters bf16: Q,K -> [B,H,T,D]; V -> transposed [B,H,D,T].
// Block-uniform facts: 64 | 2048 (split) and 64 | 128 (head), so the
// q/k/v split, head index, and d-base are constants per block.
// =====================================================================
__global__ __launch_bounds__(256) void qkv_gemm(const float* __restrict__ X,
                                                const float* __restrict__ W,
                                                bf16* __restrict__ Qo,
                                                bf16* __restrict__ Ko,
                                                bf16* __restrict__ Vt) {
  __shared__ alignas(16) bf16 As[128 * 32];
  __shared__ alignas(16) bf16 Bs[64 * 32];
  const int tid  = threadIdx.x;
  const int lane = tid & 31, wave = tid >> 5;
  const int wm = wave >> 1, wn = wave & 1;
  const int half = lane >> 4, l16 = lane & 15;
  const int bm = blockIdx.y, bn = blockIdx.x;

  const v8f zero8 = {0.f, 0.f, 0.f, 0.f, 0.f, 0.f, 0.f, 0.f};
  v8f acc[2][2] = {{zero8, zero8}, {zero8, zero8}};

  for (int k0 = 0; k0 < CD; k0 += 32) {
#pragma unroll
    for (int i = 0; i < 4; ++i) {            // A: 128x32 f32 -> bf16 LDS
      int idx = tid + i * 256;
      int row = idx >> 3, kc = (idx & 7) << 2;
      f32x4 f = *(const f32x4*)(X + (size_t)(bm * 128 + row) * CD + k0 + kc);
      *(u32x2*)(As + row * 32 + kc) = pack4_bf16(f);
    }
#pragma unroll
    for (int i = 0; i < 2; ++i) {            // B: 64x32 f32 -> bf16 LDS
      int idx = tid + i * 256;
      int row = idx >> 3, kc = (idx & 7) << 2;
      f32x4 f = *(const f32x4*)(W + (size_t)(bn * 64 + row) * CD + k0 + kc);
      *(u32x2*)(Bs + row * 32 + kc) = pack4_bf16(f);
    }
    __syncthreads();
    v16bf a0 = lds_frag(As, wm * 32 + l16,      32, lane);
    v16bf a1 = lds_frag(As, wm * 32 + 16 + l16, 32, lane);
    v16bf b0 = lds_frag(Bs, wn * 32 + l16,      32, lane);
    v16bf b1 = lds_frag(Bs, wn * 32 + 16 + l16, 32, lane);
    acc[0][0] = WMMA_BF16(a0, b0, acc[0][0]);
    acc[0][1] = WMMA_BF16(a0, b1, acc[0][1]);
    acc[1][0] = WMMA_BF16(a1, b0, acc[1][0]);
    acc[1][1] = WMMA_BF16(a1, b1, acc[1][1]);
    __syncthreads();
  }

  // ---- epilogue: block-uniform split/head/d-base, branch once ----
  const int sp    = bn >> 5;          // 0=q, 1=k, 2=v  (32 bn-blocks per split)
  const int hidx  = (bn & 31) >> 1;   // head index (block-uniform)
  const int dbase = (bn & 1) * 64;    // d offset within head (block-uniform)
  const int grow0 = bm * 128 + wm * 32 + half * 8;   // + mi*16 + r -> token bt

  if (sp < 2) {
    bf16* __restrict__ P = (sp == 0) ? Qo : Ko;
#pragma unroll
    for (int mi = 0; mi < 2; ++mi)
#pragma unroll
      for (int ni = 0; ni < 2; ++ni) {
        const int d = dbase + wn * 32 + ni * 16 + l16;
#pragma unroll
        for (int r = 0; r < 8; ++r) {
          int grow = grow0 + mi * 16 + r;
          int b = grow >> 9, t = grow & 511;
          size_t bh = (size_t)(b * HD + hidx);
          P[(bh * TD + t) * DD + d] = (bf16)acc[mi][ni][r];
        }
      }
  } else {
#pragma unroll
    for (int mi = 0; mi < 2; ++mi)
#pragma unroll
      for (int ni = 0; ni < 2; ++ni) {
        const int d = dbase + wn * 32 + ni * 16 + l16;
#pragma unroll
        for (int r = 0; r < 8; ++r) {
          int grow = grow0 + mi * 16 + r;
          int b = grow >> 9, t = grow & 511;
          size_t bh = (size_t)(b * HD + hidx);
          Vt[(bh * DD + d) * TD + t] = (bf16)acc[mi][ni][r];
        }
      }
  }
}

// =====================================================================
// Kernel 2: RoPE on Q and K (first 64 dims, pairs (j, j+32)).
// =====================================================================
__global__ __launch_bounds__(256) void rope_kernel(bf16* __restrict__ Q,
                                                   bf16* __restrict__ K) {
  int gid = blockIdx.x * 256 + threadIdx.x;   // (bh, t, j)
  int j  = gid & 31;
  int t  = (gid >> 5) & 511;
  int bh = gid >> 14;
  size_t base = ((size_t)bh * TD + t) * DD;
  float freq = __expf(-0.28782313662425576f * (float)j);  // ln(10000)/32
  float ang  = (float)t * freq;
  float c = __cosf(ang), sn = __sinf(ang);
  float q1 = (float)Q[base + j], q2 = (float)Q[base + j + 32];
  Q[base + j]      = (bf16)(q1 * c - q2 * sn);
  Q[base + j + 32] = (bf16)(q2 * c + q1 * sn);
  float k1 = (float)K[base + j], k2 = (float)K[base + j + 32];
  K[base + j]      = (bf16)(k1 * c - k2 * sn);
  K[base + j + 32] = (bf16)(k2 * c + k1 * sn);
}

// =====================================================================
// Kernel 3: causal flash attention for one (b, h, 128-row q block).
// 8 waves x 16 q rows each. Key blocks of 32 staged in LDS (K and V^T).
// QK^T: 2 n-tiles x 4 k-steps = 8 WMMA; PV: 8 d-tiles x 1 = 8 WMMA per block.
// =====================================================================
__global__ __launch_bounds__(256) void attn_kernel(const bf16* __restrict__ Q,
                                                   const bf16* __restrict__ K,
                                                   const bf16* __restrict__ Vt,
                                                   bf16* __restrict__ Y) {
  __shared__ alignas(16) bf16 Ks[32 * 128];      // [key][d]
  __shared__ alignas(16) bf16 Vs[128 * 32];      // [d][key]  (V^T tile)
  __shared__ alignas(16) bf16 Ps[8 * 16 * 32];   // per-wave P tile [qrow][key]
  const int tid = threadIdx.x, lane = tid & 31, wave = tid >> 5;
  const int half = lane >> 4, l16 = lane & 15;
  const int qb = blockIdx.x, h = blockIdx.y, b = blockIdx.z;
  const size_t bh = (size_t)(b * HD + h);
  const bf16* Qb = Q  + bh * TD * DD;
  const bf16* Kb = K  + bh * TD * DD;
  const bf16* Vb = Vt + bh * DD * TD;
  const int q0 = qb * 128 + wave * 16;          // first q row of this wave

  // resident Q fragments (4 K-steps of 32 over D=128)
  v16bf qf[4];
#pragma unroll
  for (int kk = 0; kk < 4; ++kk) {
    const bf16* p = Qb + (size_t)(q0 + l16) * DD + kk * 32 + half * 8;
    union { u32x4 u[2]; v16bf v; } r;
    r.u[0] = *(const u32x4*)p;
    r.u[1] = *(const u32x4*)(p + 16);
    qf[kk] = r.v;
  }

  const v8f zero8 = {0.f, 0.f, 0.f, 0.f, 0.f, 0.f, 0.f, 0.f};
  v8f accO[8];
  float mrow[8], lrow[8];
#pragma unroll
  for (int i = 0; i < 8; ++i) { accO[i] = zero8; mrow[i] = -1e30f; lrow[i] = 0.f; }

  const float scale = 0.08838834764831845f;     // 1/sqrt(128)
  const int nkb = (qb + 1) * 4;                 // causal: keys up to q block end

  for (int kb = 0; kb < nkb; ++kb) {
#pragma unroll
    for (int i = 0; i < 2; ++i) {               // stage K block 32x128
      int idx = tid + i * 256;
      int row = idx >> 4, ch = (idx & 15) << 3;
      *(u32x4*)(Ks + row * 128 + ch) =
          *(const u32x4*)(Kb + (size_t)(kb * 32 + row) * DD + ch);
    }
#pragma unroll
    for (int i = 0; i < 2; ++i) {               // stage V^T block 128x32
      int idx = tid + i * 256;
      int row = idx >> 2, ch = (idx & 3) << 3;
      *(u32x4*)(Vs + row * 32 + ch) =
          *(const u32x4*)(Vb + (size_t)row * TD + kb * 32 + ch);
    }
    __syncthreads();

    // S = scale * Q K^T, causal-masked
    float Sv[2][8];
#pragma unroll
    for (int nt = 0; nt < 2; ++nt) {
      v8f s = zero8;
#pragma unroll
      for (int kk = 0; kk < 4; ++kk) {
        v16bf kf = lds_frag(Ks + kk * 32, nt * 16 + l16, 128, lane);
        s = WMMA_BF16(qf[kk], kf, s);
      }
      int col = kb * 32 + nt * 16 + l16;
#pragma unroll
      for (int r = 0; r < 8; ++r) {
        int rowg = q0 + r + half * 8;
        Sv[nt][r] = (col > rowg) ? -1e30f : s[r] * scale;
      }
    }

    // online softmax (row reductions within 16-lane halves)
    float alpha[8];
#pragma unroll
    for (int r = 0; r < 8; ++r) {
      float bm2 = fmaxf(Sv[0][r], Sv[1][r]);
#pragma unroll
      for (int msk = 1; msk < 16; msk <<= 1) bm2 = fmaxf(bm2, __shfl_xor(bm2, msk, 32));
      float nm = fmaxf(mrow[r], bm2);
      alpha[r] = __expf(mrow[r] - nm);
      mrow[r] = nm;
    }
#pragma unroll
    for (int r = 0; r < 8; ++r) {
      float p0 = __expf(Sv[0][r] - mrow[r]);
      float p1 = __expf(Sv[1][r] - mrow[r]);
      Ps[wave * 512 + (r + half * 8) * 32 + l16]      = (bf16)p0;
      Ps[wave * 512 + (r + half * 8) * 32 + 16 + l16] = (bf16)p1;
      float rs = p0 + p1;
#pragma unroll
      for (int msk = 1; msk < 16; msk <<= 1) rs += __shfl_xor(rs, msk, 32);
      lrow[r] = lrow[r] * alpha[r] + rs;
    }
#pragma unroll
    for (int dt = 0; dt < 8; ++dt)
#pragma unroll
      for (int r = 0; r < 8; ++r) accO[dt][r] *= alpha[r];

    // O += P V   (A-frag of P reloaded from LDS -> layout transpose)
    v16bf pf = lds_frag(Ps + wave * 512, l16, 32, lane);
#pragma unroll
    for (int dt = 0; dt < 8; ++dt) {
      v16bf vf = lds_frag(Vs, dt * 16 + l16, 32, lane);
      accO[dt] = WMMA_BF16(pf, vf, accO[dt]);
    }
    __syncthreads();
  }

  // normalize and write Y as bf16 in [B, T, C] (= [b][t][h*128+d])
  const size_t ybase = ((size_t)b * TD + q0 + half * 8) * CD + h * DD + l16;
#pragma unroll
  for (int dt = 0; dt < 8; ++dt)
#pragma unroll
    for (int r = 0; r < 8; ++r) {
      float o = accO[dt][r] / lrow[r];
      Y[ybase + (size_t)r * CD + dt * 16] = (bf16)o;
    }
}

// =====================================================================
// Kernel 4: output projection. out[bt,f] = sum_c Y[bt,c] * w_out[f,c]
// A is already bf16 (straight LDS copy); B converted f32->bf16.
// =====================================================================
__global__ __launch_bounds__(256) void out_gemm(const bf16* __restrict__ Y,
                                                const float* __restrict__ W,
                                                float* __restrict__ Out) {
  __shared__ alignas(16) bf16 As[128 * 32];
  __shared__ alignas(16) bf16 Bs[64 * 32];
  const int tid  = threadIdx.x;
  const int lane = tid & 31, wave = tid >> 5;
  const int wm = wave >> 1, wn = wave & 1;
  const int half = lane >> 4, l16 = lane & 15;
  const int bm = blockIdx.y, bn = blockIdx.x;

  const v8f zero8 = {0.f, 0.f, 0.f, 0.f, 0.f, 0.f, 0.f, 0.f};
  v8f acc[2][2] = {{zero8, zero8}, {zero8, zero8}};

  for (int k0 = 0; k0 < CD; k0 += 32) {
#pragma unroll
    for (int i = 0; i < 2; ++i) {             // A: 128x32 bf16 copy
      int idx = tid + i * 256;
      int row = idx >> 2, ch = (idx & 3) << 3;
      *(u32x4*)(As + row * 32 + ch) =
          *(const u32x4*)(Y + (size_t)(bm * 128 + row) * CD + k0 + ch);
    }
#pragma unroll
    for (int i = 0; i < 2; ++i) {             // B: 64x32 f32 -> bf16
      int idx = tid + i * 256;
      int row = idx >> 3, kc = (idx & 7) << 2;
      f32x4 f = *(const f32x4*)(W + (size_t)(bn * 64 + row) * CD + k0 + kc);
      *(u32x2*)(Bs + row * 32 + kc) = pack4_bf16(f);
    }
    __syncthreads();
    v16bf a0 = lds_frag(As, wm * 32 + l16,      32, lane);
    v16bf a1 = lds_frag(As, wm * 32 + 16 + l16, 32, lane);
    v16bf b0 = lds_frag(Bs, wn * 32 + l16,      32, lane);
    v16bf b1 = lds_frag(Bs, wn * 32 + 16 + l16, 32, lane);
    acc[0][0] = WMMA_BF16(a0, b0, acc[0][0]);
    acc[0][1] = WMMA_BF16(a0, b1, acc[0][1]);
    acc[1][0] = WMMA_BF16(a1, b0, acc[1][0]);
    acc[1][1] = WMMA_BF16(a1, b1, acc[1][1]);
    __syncthreads();
  }

  const int gcol0 = bn * 64 + wn * 32 + l16;
  const int grow0 = bm * 128 + wm * 32 + half * 8;
#pragma unroll
  for (int mi = 0; mi < 2; ++mi)
#pragma unroll
    for (int ni = 0; ni < 2; ++ni)
#pragma unroll
      for (int r = 0; r < 8; ++r)
        Out[(size_t)(grow0 + mi * 16 + r) * CD + gcol0 + ni * 16] = acc[mi][ni][r];
}

// =====================================================================
extern "C" void kernel_launch(void* const* d_in, const int* in_sizes, int n_in,
                              void* d_out, int out_size, void* d_ws, size_t ws_size,
                              hipStream_t stream) {
  (void)in_sizes; (void)n_in; (void)out_size; (void)ws_size;
  const float* x     = (const float*)d_in[0];
  const float* w_qkv = (const float*)d_in[1];
  const float* w_out = (const float*)d_in[2];
  float* out = (float*)d_out;

  const size_t QKV_ELEMS = (size_t)BD * HD * TD * DD;  // 33,554,432
  bf16* Q  = (bf16*)d_ws;
  bf16* K  = Q  + QKV_ELEMS;
  bf16* Vt = K  + QKV_ELEMS;
  bf16* Y  = Vt + QKV_ELEMS;

  // qkv: M=16384, N=6144 -> grid (6144/64, 16384/128)
  qkv_gemm<<<dim3(96, 128), 256, 0, stream>>>(x, w_qkv, Q, K, Vt);
  // rope: B*H*T*32 pairs
  rope_kernel<<<(BD * HD * TD * 32) / 256, 256, 0, stream>>>(Q, K);
  // attention: (q-block, h, b)
  attn_kernel<<<dim3(TD / 128, HD, BD), 256, 0, stream>>>(Q, K, Vt, Y);
  // out proj: M=16384, N=2048 -> grid (2048/64, 16384/128)
  out_gemm<<<dim3(32, 128), 256, 0, stream>>>(Y, w_out, out);
}
